// SelfAttention_84585085927797
// MI455X (gfx1250) — matrix operand order repaired and checked by reading, not compiled, not run on hardware
//
#include <hip/hip_runtime.h>
#include <math.h>

#define B_ 8
#define C_ 512
#define T_ 2048
#define E_ 512
#define LN_EPS 1e-5f
#define LDSROW 40   // 32 f16 + 8 pad, keeps 16B alignment, breaks bank conflicts

typedef __attribute__((ext_vector_type(16))) _Float16 v16h;
typedef __attribute__((ext_vector_type(8)))  float    v8f;

union FragU {
    v16h v;
    uint4 q[2];
    _Float16 h[16];
};

// ---------------------------------------------------------------------------
// WMMA fragment load + MMA for one BK=32 step.
// A fragment (16x32 f16): lane L (half = L>>4) holds row M=L&15,
//   K = half*8 .. half*8+7  and  K = 16+half*8 .. 16+half*8+7  (ISA 7.12.2).
// B fragment (32x16 f16): lane L holds col N=L&15, same K layout as A.
// C/D (16x16 f32): VGPR r, lane L -> (M = r + 8*(L>>4), N = L&15).
// ---------------------------------------------------------------------------
__device__ __forceinline__ void frag_mma_step(const _Float16* As, const _Float16* Bs,
                                              v8f (&acc)[2][4], int lane, int wr, int wc) {
    FragU af[2], bf[4];
    const int m16  = lane & 15;
    const int kb8  = (lane >> 4) << 3;
    const int kb16 = (lane >> 4) << 4;
#pragma unroll
    for (int i = 0; i < 2; ++i) {
        const _Float16* p = As + (wr * 32 + i * 16 + m16) * LDSROW + kb8;
        af[i].q[0] = *(const uint4*)(p);        // K = kb8 .. kb8+7
        af[i].q[1] = *(const uint4*)(p + 16);   // K = kb8+16 .. kb8+23
    }
#pragma unroll
    for (int j = 0; j < 4; ++j) {
        const _Float16* p = Bs + (wc * 64 + j * 16 + m16) * LDSROW + kb8;
        bf[j].q[0] = *(const uint4*)(p);        // K = kb8 .. kb8+7
        bf[j].q[1] = *(const uint4*)(p + 16);   // K = kb8+16 .. kb8+23
    }
#pragma unroll
    for (int i = 0; i < 2; ++i)
#pragma unroll
        for (int j = 0; j < 4; ++j)
            acc[i][j] = __builtin_amdgcn_wmma_f32_16x16x32_f16(
                false, af[i].v, false, bf[j].v, (short)0, acc[i][j], false, false);
}

__device__ __forceinline__ void zero_acc(v8f (&acc)[2][4]) {
#pragma unroll
    for (int i = 0; i < 2; ++i)
#pragma unroll
        for (int j = 0; j < 4; ++j)
#pragma unroll
            for (int r = 0; r < 8; ++r) acc[i][j][r] = 0.0f;
}

// Generic f16 GEMM main loop: D(128x128) += A[M,K] * Bt[N,K]^T, LDS staged.
__device__ __forceinline__ void gemm_loop_f16(const _Float16* A, int lda,
                                              const _Float16* Bt, int ldb, int K,
                                              int m0blk, int n0blk,
                                              _Float16* As, _Float16* Bs,
                                              v8f (&acc)[2][4]) {
    const int tid  = threadIdx.x;
    const int lane = tid & 31;
    const int wr   = (tid >> 5) >> 1;   // 0..3 : 32-row band
    const int wc   = (tid >> 5) & 1;    // 0..1 : 64-col band
    const int srow = tid >> 1;          // 0..127
    const int scol = (tid & 1) << 4;    // 0 or 16
    for (int k0 = 0; k0 < K; k0 += 32) {
        const _Float16* ga = A  + (size_t)(m0blk + srow) * lda + k0 + scol;
        const _Float16* gb = Bt + (size_t)(n0blk + srow) * ldb + k0 + scol;
        uint4 a0 = *(const uint4*)(ga);
        uint4 a1 = *(const uint4*)(ga + 8);
        uint4 b0 = *(const uint4*)(gb);
        uint4 b1 = *(const uint4*)(gb + 8);
        __syncthreads();
        *(uint4*)(As + srow * LDSROW + scol)     = a0;
        *(uint4*)(As + srow * LDSROW + scol + 8) = a1;
        *(uint4*)(Bs + srow * LDSROW + scol)     = b0;
        *(uint4*)(Bs + srow * LDSROW + scol + 8) = b1;
        __syncthreads();
        frag_mma_step(As, Bs, acc, lane, wr, wc);
    }
}

// ---------------------------------------------------------------------------
// Transpose + cast kernels
// ---------------------------------------------------------------------------

// x[B,C,T] f32 -> residual[B,T,C] f16
__global__ __launch_bounds__(256) void xpose_x_kernel(const float* __restrict__ x,
                                                      _Float16* __restrict__ res) {
    __shared__ float tile[32][33];
    const int t0 = blockIdx.x * 32, c0 = blockIdx.y * 32, b = blockIdx.z;
    const int tx = threadIdx.x & 31, ty = threadIdx.x >> 5;  // ty: 0..7
    const float* xb = x + (size_t)b * C_ * T_;
#pragma unroll
    for (int i = 0; i < 4; ++i)
        tile[ty + 8 * i][tx] = xb[(size_t)(c0 + ty + 8 * i) * T_ + t0 + tx];
    __syncthreads();
    _Float16* rb = res + (size_t)b * T_ * C_;
#pragma unroll
    for (int i = 0; i < 4; ++i)
        rb[(size_t)(t0 + ty + 8 * i) * C_ + c0 + tx] = (_Float16)tile[tx][ty + 8 * i];
}

// dst[j,i] = (f16)src[i,j] for 512x512 matrices; z selects Wq/Wk/Wv/Wo
__global__ __launch_bounds__(256) void xpose_w_kernel(const float* __restrict__ Wq,
                                                      const float* __restrict__ Wk,
                                                      const float* __restrict__ Wv,
                                                      const float* __restrict__ Wo,
                                                      _Float16* __restrict__ wqt,
                                                      _Float16* __restrict__ wkt,
                                                      _Float16* __restrict__ wvt,
                                                      _Float16* __restrict__ wot) {
    __shared__ float tile[32][33];
    const int z = blockIdx.z;
    const float* src = (z == 0) ? Wq : (z == 1) ? Wk : (z == 2) ? Wv : Wo;
    _Float16*   dst = (z == 0) ? wqt : (z == 1) ? wkt : (z == 2) ? wvt : wot;
    const int i0 = blockIdx.x * 32, j0 = blockIdx.y * 32;
    const int tx = threadIdx.x & 31, ty = threadIdx.x >> 5;
#pragma unroll
    for (int i = 0; i < 4; ++i)
        tile[ty + 8 * i][tx] = src[(size_t)(i0 + ty + 8 * i) * 512 + j0 + tx];
    __syncthreads();
#pragma unroll
    for (int i = 0; i < 4; ++i)
        dst[(size_t)(j0 + ty + 8 * i) * 512 + i0 + tx] = (_Float16)tile[tx][ty + 8 * i];
}

// ---------------------------------------------------------------------------
// GEMM 1: q/k/v = residual @ W + b   (grid.z = b*3 + which)
// ---------------------------------------------------------------------------
__global__ __launch_bounds__(256) void qkv_gemm_kernel(const _Float16* __restrict__ res,
                                                       const _Float16* __restrict__ wqt,
                                                       const _Float16* __restrict__ wkt,
                                                       const _Float16* __restrict__ wvt,
                                                       const float* __restrict__ bq,
                                                       const float* __restrict__ bk,
                                                       const float* __restrict__ bv,
                                                       _Float16* __restrict__ q,
                                                       _Float16* __restrict__ k,
                                                       _Float16* __restrict__ v) {
    __shared__ _Float16 As[128 * LDSROW];
    __shared__ _Float16 Bs[128 * LDSROW];
    const int zb = blockIdx.z, b = zb / 3, which = zb % 3;
    const _Float16* A  = res + (size_t)b * T_ * C_;
    const _Float16* Bt = (which == 0) ? wqt : (which == 1) ? wkt : wvt;
    const float* bias  = (which == 0) ? bq : (which == 1) ? bk : bv;
    _Float16* out = ((which == 0) ? q : (which == 1) ? k : v) + (size_t)b * T_ * E_;
    const int n0blk = blockIdx.x * 128, m0blk = blockIdx.y * 128;
    v8f acc[2][4];
    zero_acc(acc);
    gemm_loop_f16(A, C_, Bt, C_, C_, m0blk, n0blk, As, Bs, acc);
    const int lane = threadIdx.x & 31, wave = threadIdx.x >> 5;
    const int mB = m0blk + (wave >> 1) * 32, nB = n0blk + (wave & 1) * 64;
#pragma unroll
    for (int j = 0; j < 4; ++j) {
        const int n = nB + j * 16 + (lane & 15);
        const float bval = bias[n];
#pragma unroll
        for (int i = 0; i < 2; ++i) {
            const int m0 = mB + i * 16 + ((lane >> 4) << 3);
#pragma unroll
            for (int r = 0; r < 8; ++r)
                out[(size_t)(m0 + r) * E_ + n] = (_Float16)(acc[i][j][r] + bval);
        }
    }
}

// ---------------------------------------------------------------------------
// GEMM 2: energy[b,t,s] = q[b,t,:] . k[b,s,:]  (raw scores, f32, into d_out)
// ---------------------------------------------------------------------------
__global__ __launch_bounds__(256) void energy_kernel(const _Float16* __restrict__ q,
                                                     const _Float16* __restrict__ k,
                                                     float* __restrict__ attn) {
    __shared__ _Float16 As[128 * LDSROW];
    __shared__ _Float16 Bs[128 * LDSROW];
    const int b = blockIdx.z;
    const _Float16* A  = q + (size_t)b * T_ * E_;
    const _Float16* Bt = k + (size_t)b * T_ * E_;   // K row-major == Bt layout
    float* out = attn + (size_t)b * T_ * T_;
    const int n0blk = blockIdx.x * 128, m0blk = blockIdx.y * 128;
    v8f acc[2][4];
    zero_acc(acc);
    gemm_loop_f16(A, E_, Bt, E_, E_, m0blk, n0blk, As, Bs, acc);
    const int lane = threadIdx.x & 31, wave = threadIdx.x >> 5;
    const int mB = m0blk + (wave >> 1) * 32, nB = n0blk + (wave & 1) * 64;
#pragma unroll
    for (int j = 0; j < 4; ++j) {
        const int n = nB + j * 16 + (lane & 15);
#pragma unroll
        for (int i = 0; i < 2; ++i) {
            const int m0 = mB + i * 16 + ((lane >> 4) << 3);
#pragma unroll
            for (int r = 0; r < 8; ++r)
                out[(size_t)(m0 + r) * T_ + n] = acc[i][j][r];
        }
    }
}

// ---------------------------------------------------------------------------
// In-place fp32 row softmax over T_ elements (one block per row)
// ---------------------------------------------------------------------------
__global__ __launch_bounds__(256) void softmax_kernel(float* __restrict__ attn) {
    __shared__ float red[256];
    const int tid = threadIdx.x;
    float* p = attn + (size_t)blockIdx.x * T_;
    float v[8];
    float mx = -3.402823e38f;
#pragma unroll
    for (int i = 0; i < 8; ++i) { v[i] = p[tid + (i << 8)]; mx = fmaxf(mx, v[i]); }
    red[tid] = mx; __syncthreads();
    for (int s = 128; s > 0; s >>= 1) {
        if (tid < s) red[tid] = fmaxf(red[tid], red[tid + s]);
        __syncthreads();
    }
    mx = red[0]; __syncthreads();
    float sum = 0.f;
#pragma unroll
    for (int i = 0; i < 8; ++i) { v[i] = expf(v[i] - mx); sum += v[i]; }
    red[tid] = sum; __syncthreads();
    for (int s = 128; s > 0; s >>= 1) {
        if (tid < s) red[tid] += red[tid + s];
        __syncthreads();
    }
    const float inv = 1.0f / red[0];
#pragma unroll
    for (int i = 0; i < 8; ++i) p[tid + (i << 8)] = v[i] * inv;
}

// ---------------------------------------------------------------------------
// GEMM 3: attended = attn(f32, from d_out) @ V.  A converted f32->f16 in
// staging; V block is transposed during LDS staging to get Bt layout.
// ---------------------------------------------------------------------------
__global__ __launch_bounds__(256) void attend_kernel(const float* __restrict__ attn,
                                                     const _Float16* __restrict__ v,
                                                     _Float16* __restrict__ att) {
    __shared__ _Float16 As[128 * LDSROW];
    __shared__ _Float16 Bs[128 * LDSROW];
    const int b = blockIdx.z;
    const float* Aat = attn + (size_t)b * T_ * T_;
    const _Float16* Vb = v + (size_t)b * T_ * E_;
    _Float16* out = att + (size_t)b * T_ * E_;
    const int n0blk = blockIdx.x * 128, m0blk = blockIdx.y * 128;
    const int tid = threadIdx.x;
    const int lane = tid & 31;
    const int wr = (tid >> 5) >> 1, wc = (tid >> 5) & 1;
    const int srow = tid >> 1, scol = (tid & 1) << 4;   // A staging: 128 x 32
    const int vs = tid >> 3, ve = (tid & 7) << 4;       // B staging: 32 x 128
    v8f acc[2][4];
    zero_acc(acc);
    for (int k0 = 0; k0 < T_; k0 += 32) {
        // A: fp32 attn tile -> f16
        const float* ga = Aat + (size_t)(m0blk + srow) * T_ + k0 + scol;
        float4 f0 = *(const float4*)(ga + 0);
        float4 f1 = *(const float4*)(ga + 4);
        float4 f2 = *(const float4*)(ga + 8);
        float4 f3 = *(const float4*)(ga + 12);
        FragU hp;
        hp.h[0]  = (_Float16)f0.x; hp.h[1]  = (_Float16)f0.y;
        hp.h[2]  = (_Float16)f0.z; hp.h[3]  = (_Float16)f0.w;
        hp.h[4]  = (_Float16)f1.x; hp.h[5]  = (_Float16)f1.y;
        hp.h[6]  = (_Float16)f1.z; hp.h[7]  = (_Float16)f1.w;
        hp.h[8]  = (_Float16)f2.x; hp.h[9]  = (_Float16)f2.y;
        hp.h[10] = (_Float16)f2.z; hp.h[11] = (_Float16)f2.w;
        hp.h[12] = (_Float16)f3.x; hp.h[13] = (_Float16)f3.y;
        hp.h[14] = (_Float16)f3.z; hp.h[15] = (_Float16)f3.w;
        // B: V rows k0+vs, cols n0blk+ve .. +15 (coalesced read)
        const _Float16* gv = Vb + (size_t)(k0 + vs) * E_ + n0blk + ve;
        FragU vp;
        vp.q[0] = *(const uint4*)(gv);
        vp.q[1] = *(const uint4*)(gv + 8);
        __syncthreads();
        *(uint4*)(As + srow * LDSROW + scol)     = hp.q[0];
        *(uint4*)(As + srow * LDSROW + scol + 8) = hp.q[1];
#pragma unroll
        for (int i = 0; i < 16; ++i)            // transpose into Bt layout
            Bs[(ve + i) * LDSROW + vs] = vp.h[i];
        __syncthreads();
        frag_mma_step(As, Bs, acc, lane, wr, wc);
    }
    const int mB = m0blk + wr * 32, nB = n0blk + wc * 64;
#pragma unroll
    for (int j = 0; j < 4; ++j) {
        const int n = nB + j * 16 + (lane & 15);
#pragma unroll
        for (int i = 0; i < 2; ++i) {
            const int m0 = mB + i * 16 + ((lane >> 4) << 3);
#pragma unroll
            for (int r = 0; r < 8; ++r)
                out[(size_t)(m0 + r) * E_ + n] = (_Float16)acc[i][j][r];
        }
    }
}

// ---------------------------------------------------------------------------
// GEMM 4: h = attended @ Wo + bo + residual(x^T), f32 out
// ---------------------------------------------------------------------------
__global__ __launch_bounds__(256) void outproj_kernel(const _Float16* __restrict__ att,
                                                      const _Float16* __restrict__ wot,
                                                      const float* __restrict__ bo,
                                                      const float* __restrict__ x,
                                                      float* __restrict__ hbuf) {
    __shared__ _Float16 As[128 * LDSROW];
    __shared__ _Float16 Bs[128 * LDSROW];
    const int b = blockIdx.z;
    const _Float16* A = att + (size_t)b * T_ * E_;
    const float* xb = x + (size_t)b * C_ * T_;
    float* hB = hbuf + (size_t)b * T_ * C_;
    const int n0blk = blockIdx.x * 128, m0blk = blockIdx.y * 128;
    v8f acc[2][4];
    zero_acc(acc);
    gemm_loop_f16(A, E_, wot, E_, E_, m0blk, n0blk, As, Bs, acc);
    const int lane = threadIdx.x & 31, wave = threadIdx.x >> 5;
    const int mB = m0blk + (wave >> 1) * 32, nB = n0blk + (wave & 1) * 64;
#pragma unroll
    for (int j = 0; j < 4; ++j) {
        const int n = nB + j * 16 + (lane & 15);
        const float bval = bo[n];
#pragma unroll
        for (int i = 0; i < 2; ++i) {
            const int m0 = mB + i * 16 + ((lane >> 4) << 3);
#pragma unroll
            for (int r = 0; r < 8; ++r) {
                const int m = m0 + r;
                hB[(size_t)m * C_ + n] = acc[i][j][r] + bval + xb[(size_t)n * T_ + m];
            }
        }
    }
}

// ---------------------------------------------------------------------------
// fp32 LayerNorm over C_ (one block per (b,t) row), writes final output
// ---------------------------------------------------------------------------
__global__ __launch_bounds__(256) void layernorm_kernel(const float* __restrict__ hbuf,
                                                        const float* __restrict__ gamma,
                                                        const float* __restrict__ beta,
                                                        float* __restrict__ out) {
    __shared__ float red[256];
    const int tid = threadIdx.x;
    const float* hp = hbuf + (size_t)blockIdx.x * C_;
    const float a0 = hp[tid], a1 = hp[tid + 256];
    red[tid] = a0 + a1; __syncthreads();
    for (int s = 128; s > 0; s >>= 1) {
        if (tid < s) red[tid] += red[tid + s];
        __syncthreads();
    }
    const float mu = red[0] * (1.0f / C_); __syncthreads();
    const float d0 = a0 - mu, d1 = a1 - mu;
    red[tid] = d0 * d0 + d1 * d1; __syncthreads();
    for (int s = 128; s > 0; s >>= 1) {
        if (tid < s) red[tid] += red[tid + s];
        __syncthreads();
    }
    const float rstd = rsqrtf(red[0] * (1.0f / C_) + LN_EPS);
    float* op = out + (size_t)blockIdx.x * C_;
    op[tid]       = d0 * rstd * gamma[tid] + beta[tid];
    op[tid + 256] = d1 * rstd * gamma[tid + 256] + beta[tid + 256];
}

// ---------------------------------------------------------------------------
extern "C" void kernel_launch(void* const* d_in, const int* in_sizes, int n_in,
                              void* d_out, int out_size, void* d_ws, size_t ws_size,
                              hipStream_t stream) {
    const float* x     = (const float*)d_in[0];
    const float* Wq    = (const float*)d_in[1];
    const float* bq    = (const float*)d_in[2];
    const float* Wk    = (const float*)d_in[3];
    const float* bk    = (const float*)d_in[4];
    const float* Wv    = (const float*)d_in[5];
    const float* bv    = (const float*)d_in[6];
    const float* Wo    = (const float*)d_in[7];
    const float* bo    = (const float*)d_in[8];
    const float* gamma = (const float*)d_in[9];
    const float* beta  = (const float*)d_in[10];

    float* attnOut = (float*)d_out;                          // [B,T,T]
    float* lnOut   = attnOut + (size_t)B_ * T_ * T_;         // [B,T,C]

    char* ws = (char*)d_ws;
    _Float16* resb = (_Float16*)ws;  ws += (size_t)B_ * T_ * C_ * sizeof(_Float16);
    _Float16* wqt  = (_Float16*)ws;  ws += (size_t)E_ * C_ * sizeof(_Float16);
    _Float16* wkt  = (_Float16*)ws;  ws += (size_t)E_ * C_ * sizeof(_Float16);
    _Float16* wvt  = (_Float16*)ws;  ws += (size_t)E_ * C_ * sizeof(_Float16);
    _Float16* wot  = (_Float16*)ws;  ws += (size_t)C_ * E_ * sizeof(_Float16);
    _Float16* qb   = (_Float16*)ws;  ws += (size_t)B_ * T_ * E_ * sizeof(_Float16);
    _Float16* kb   = (_Float16*)ws;  ws += (size_t)B_ * T_ * E_ * sizeof(_Float16);
    _Float16* vb   = (_Float16*)ws;  ws += (size_t)B_ * T_ * E_ * sizeof(_Float16);
    _Float16* attb = (_Float16*)ws;  ws += (size_t)B_ * T_ * E_ * sizeof(_Float16);
    float*    hbuf = (float*)ws;     ws += (size_t)B_ * T_ * C_ * sizeof(float);

    // 1. residual (f16, [B,T,C]) and transposed/cast weights
    xpose_x_kernel<<<dim3(T_ / 32, C_ / 32, B_), 256, 0, stream>>>(x, resb);
    xpose_w_kernel<<<dim3(16, 16, 4), 256, 0, stream>>>(Wq, Wk, Wv, Wo, wqt, wkt, wvt, wot);
    // 2. q/k/v projections (WMMA)
    qkv_gemm_kernel<<<dim3(E_ / 128, T_ / 128, B_ * 3), 256, 0, stream>>>(
        resb, wqt, wkt, wvt, bq, bk, bv, qb, kb, vb);
    // 3. energy = q @ k^T (raw scores into d_out)
    energy_kernel<<<dim3(T_ / 128, T_ / 128, B_), 256, 0, stream>>>(qb, kb, attnOut);
    // 4. softmax in place (fp32)
    softmax_kernel<<<dim3(B_ * T_), 256, 0, stream>>>(attnOut);
    // 5. attended = attn @ v (WMMA, attn converted f32->f16 in staging)
    attend_kernel<<<dim3(E_ / 128, T_ / 128, B_), 256, 0, stream>>>(attnOut, vb, attb);
    // 6. out projection + bias + fp32 residual
    outproj_kernel<<<dim3(C_ / 128, T_ / 128, B_), 256, 0, stream>>>(attb, wot, bo, x, hbuf);
    // 7. LayerNorm -> final output
    layernorm_kernel<<<dim3(B_ * T_), 256, 0, stream>>>(hbuf, gamma, beta, lnOut);
}
